// AnomalyDetectionGNN_58789512348245
// MI455X (gfx1250) — matrix operand order repaired and checked by reading, not compiled, not run on hardware
//
#include <hip/hip_runtime.h>
#include <hip/hip_bf16.h>

// ---------------------------------------------------------------------------
// AnomalyDetection GNN (3x GAT + LN + scorer MLP) for MI455X (gfx1250, wave32)
// Dense GEMMs run on v_wmma_f32_16x16x32_f16 (f16 inputs, fp32 accumulate),
// 2 WMMA tiles per wave per K-step. Irregular edge softmax/aggregation uses
// fp32 global atomics (accumulator tensors are L2-resident: 80MB < 192MB L2).
// ---------------------------------------------------------------------------

typedef __attribute__((ext_vector_type(16))) _Float16 v16h;
typedef __attribute__((ext_vector_type(8)))  _Float16 v8h;
typedef __attribute__((ext_vector_type(8)))  float    v8f;

#define NN 20000
#define EE 320000
#define DD 512

// ---------------------------------------------------------------------------
// Zero fill
// ---------------------------------------------------------------------------
__global__ void kfill0(unsigned* __restrict__ p, long n) {
  long i = (long)blockIdx.x * blockDim.x + threadIdx.x;
  if (i < n) p[i] = 0u;
}

// ---------------------------------------------------------------------------
// WMMA GEMM: C[M,Nc] = A[M,K] * B[K,Nc]   (fp32 in/out, f16 multiply)
// block = 256 threads (8 waves). Block tile 32x128; wave tile 16x32
// (one A fragment, two B fragments, two accumulators -> 2 v_wmma / K-step).
// Requires M % 32 == 0, K % 32 == 0. Nc guarded (partial col tiles OK).
// ---------------------------------------------------------------------------
#define TM 32
#define TNB 128
#define TK 32
#define LDSP 48   // LDS row stride in halves (96B; 16B-aligned sub-chunks)

__global__ __launch_bounds__(256)
void gemm_wmma_f16(const float* __restrict__ A, const float* __restrict__ B,
                   float* __restrict__ C, int M, int K, int Nc) {
  __shared__ _Float16 As[TM][LDSP];
  __shared__ _Float16 Bs[TNB][LDSP];

  const int t    = threadIdx.x;
  const int lane = t & 31;
  const int wv   = t >> 5;
  const int wr   = wv & 1;   // 0..1 : 16-row sub-tile
  const int wc   = wv >> 1;  // 0..3 : 32-col group
  const int rowBase = blockIdx.x * TM;
  const int colBase = blockIdx.y * TNB;
  const bool fullN  = (colBase + TNB) <= Nc;   // block-uniform

  v8f c0 = {}, c1 = {};

  for (int kb = 0; kb < K; kb += TK) {
    // ---- load A tile (32 rows x 32 K): 1 float4 / thread ----
    {
      const int r  = t >> 3;
      const int kg = (t & 7) * 4;
      const float* ap = A + (size_t)(rowBase + r) * K + kb + kg;
      const float4 va = *(const float4*)ap;
      As[r][kg + 0] = (_Float16)va.x;
      As[r][kg + 1] = (_Float16)va.y;
      As[r][kg + 2] = (_Float16)va.z;
      As[r][kg + 3] = (_Float16)va.w;
      if (kb + TK < K) __builtin_prefetch(ap + TK, 0, 3);
    }
    // ---- load B tile (32 K x 128 N): 4 float4 / thread, K-contig in LDS ----
    if (fullN) {
      #pragma unroll
      for (int g = 0; g < 4; ++g) {
        const int idx = t + 256 * g;        // 1024 float4 groups
        const int k   = idx >> 5;           // 0..31
        const int nq  = (idx & 31) * 4;     // 0..124
        const float* bp = B + (size_t)(kb + k) * Nc + colBase + nq;
        const float4 vb = *(const float4*)bp;
        Bs[nq + 0][k] = (_Float16)vb.x;
        Bs[nq + 1][k] = (_Float16)vb.y;
        Bs[nq + 2][k] = (_Float16)vb.z;
        Bs[nq + 3][k] = (_Float16)vb.w;
        if (kb + TK < K) __builtin_prefetch(bp + (size_t)TK * Nc, 0, 3);
      }
    } else {
      #pragma unroll
      for (int g = 0; g < 4; ++g) {
        const int idx = t + 256 * g;
        const int k   = idx >> 5;
        const int nq  = (idx & 31) * 4;
        const int col = colBase + nq;
        const size_t rowoff = (size_t)(kb + k) * Nc;
        Bs[nq + 0][k] = (col + 0 < Nc) ? (_Float16)B[rowoff + col + 0] : (_Float16)0.f;
        Bs[nq + 1][k] = (col + 1 < Nc) ? (_Float16)B[rowoff + col + 1] : (_Float16)0.f;
        Bs[nq + 2][k] = (col + 2 < Nc) ? (_Float16)B[rowoff + col + 2] : (_Float16)0.f;
        Bs[nq + 3][k] = (col + 3 < Nc) ? (_Float16)B[rowoff + col + 3] : (_Float16)0.f;
      }
    }
    __syncthreads();

    // ---- fragments in contiguous k order (k = 16*half + s), same for A and B ----
    const int half  = lane >> 4;             // 0: lanes 0-15, 1: lanes 16-31
    const int mrow  = wr * 16 + (lane & 15);
    const int ncol0 = wc * 32 + (lane & 15);
    const int ncol1 = ncol0 + 16;

    union { v16h v; v8h h[2]; } ua, ub0, ub1;
    // A 16x32: lanes<16 -> K 0..15, lanes>=16 -> K 16..31, row = lane&15
    ua.h[0]  = *(const v8h*)&As[mrow][half * 16];
    ua.h[1]  = *(const v8h*)&As[mrow][half * 16 + 8];
    // B 32x16: lanes<16 -> K 0..15, lanes>=16 -> K 16..31, col = lane&15
    ub0.h[0] = *(const v8h*)&Bs[ncol0][half * 16];
    ub0.h[1] = *(const v8h*)&Bs[ncol0][half * 16 + 8];
    ub1.h[0] = *(const v8h*)&Bs[ncol1][half * 16];
    ub1.h[1] = *(const v8h*)&Bs[ncol1][half * 16 + 8];

    c0 = __builtin_amdgcn_wmma_f32_16x16x32_f16(false, ua.v, false, ub0.v,
                                                (short)0, c0, false, false);
    c1 = __builtin_amdgcn_wmma_f32_16x16x32_f16(false, ua.v, false, ub1.v,
                                                (short)0, c1, false, false);
    __syncthreads();
  }

  // ---- store: VGPR r -> M = r + (half?8:0), N = lane&15 ----
  const int rbase = rowBase + wr * 16 + ((lane >> 4) * 8);
  const int n0    = colBase + wc * 32 + (lane & 15);
  const int n1    = n0 + 16;
  if (n0 < Nc) {
    #pragma unroll
    for (int r = 0; r < 8; ++r)
      C[(size_t)(rbase + r) * Nc + n0] = c0[r];
  }
  if (n1 < Nc) {
    #pragma unroll
    for (int r = 0; r < 8; ++r)
      C[(size_t)(rbase + r) * Nc + n1] = c1[r];
  }
}

// ---------------------------------------------------------------------------
// Attention scores: es[n,h] = sum_c h[n,h,c]*a_src[h,c]; ed analogous.
// grid = N, block = H*32 (one wave per head).
// ---------------------------------------------------------------------------
__global__ void attn_scores(const float* __restrict__ h,
                            const float* __restrict__ a_src,
                            const float* __restrict__ a_dst,
                            float* __restrict__ es, float* __restrict__ ed,
                            int H, int C) {
  int n = blockIdx.x;
  int wv = threadIdx.x >> 5, lane = threadIdx.x & 31;
  if (wv >= H) return;
  const float* hp = h + (size_t)n * H * C + (size_t)wv * C;
  float ss = 0.f, sd = 0.f;
  for (int c = lane; c < C; c += 32) {
    float v = hp[c];
    ss += v * a_src[wv * C + c];
    sd += v * a_dst[wv * C + c];
  }
  for (int off = 16; off; off >>= 1) {
    ss += __shfl_down(ss, off, 32);
    sd += __shfl_down(sd, off, 32);
  }
  if (lane == 0) { es[n * H + wv] = ss; ed[n * H + wv] = sd; }
}

// ---------------------------------------------------------------------------
// Monotone uint key for float atomicMax
// ---------------------------------------------------------------------------
__device__ __forceinline__ unsigned fkey(float f) {
  unsigned u = __float_as_uint(f);
  return (u & 0x80000000u) ? ~u : (u | 0x80000000u);
}
__device__ __forceinline__ float funkey(unsigned k) {
  return (k & 0x80000000u) ? __uint_as_float(k & 0x7FFFFFFFu)
                           : __uint_as_float(~k);
}

// pass 1: e = leaky_relu(es[src]+ed[dst]); atomicMax segment max per dst
__global__ void edge_pass1(const int* __restrict__ ei, int E, int N,
                           const float* __restrict__ es, const float* __restrict__ ed,
                           float* __restrict__ ebuf, unsigned* __restrict__ mkey, int H) {
  int e = blockIdx.x * blockDim.x + threadIdx.x;
  if (e >= E + N) return;
  int s = (e < E) ? ei[e]     : (e - E);
  int d = (e < E) ? ei[E + e] : (e - E);
  for (int h = 0; h < H; ++h) {
    float v = es[s * H + h] + ed[d * H + h];
    v = (v > 0.f) ? v : 0.2f * v;
    ebuf[(size_t)e * H + h] = v;
    atomicMax(&mkey[d * H + h], fkey(v));
  }
}

// pass 2: ex = exp(e - m[dst]); z[dst] += ex
__global__ void edge_pass2(const int* __restrict__ ei, int E, int N,
                           float* __restrict__ ebuf, const unsigned* __restrict__ mkey,
                           float* __restrict__ z, int H) {
  int e = blockIdx.x * blockDim.x + threadIdx.x;
  if (e >= E + N) return;
  int d = (e < E) ? ei[E + e] : (e - E);
  for (int h = 0; h < H; ++h) {
    float m  = funkey(mkey[d * H + h]);
    float ex = __expf(ebuf[(size_t)e * H + h] - m);
    ebuf[(size_t)e * H + h] = ex;
    atomicAdd(&z[d * H + h], ex);
  }
}

// pass 2b: alpha = ex / (z[dst] + eps)
__global__ void edge_alpha(const int* __restrict__ ei, int E, int N,
                           const float* __restrict__ z, float* __restrict__ ebuf, int H) {
  int e = blockIdx.x * blockDim.x + threadIdx.x;
  if (e >= E + N) return;
  int d = (e < E) ? ei[E + e] : (e - E);
  for (int h = 0; h < H; ++h)
    ebuf[(size_t)e * H + h] /= (z[d * H + h] + 1e-16f);
}

// pass 3: agg[dst, f] += h[src, f] * alpha[e, f>>cshift]
__global__ void edge_agg(const int* __restrict__ ei, int E, int N,
                         const float* __restrict__ alpha, const float* __restrict__ h,
                         float* __restrict__ agg, int H, int cshift, int fshift) {
  size_t idx   = (size_t)blockIdx.x * blockDim.x + threadIdx.x;
  size_t total = (size_t)(E + N) << fshift;
  if (idx >= total) return;
  int e = (int)(idx >> fshift);
  int f = (int)(idx & (((size_t)1 << fshift) - 1));
  int s = (e < E) ? ei[e]     : (e - E);
  int d = (e < E) ? ei[E + e] : (e - E);
  float a = alpha[(size_t)e * H + (f >> cshift)];
  atomicAdd(&agg[((size_t)d << fshift) + f], h[((size_t)s << fshift) + f] * a);
}

// ---------------------------------------------------------------------------
// out = LN(relu(in + bias)) * g + be  (+ optional residual)
// grid = N, block = min(F,256), perThread = F/blockDim (<=4)
// ---------------------------------------------------------------------------
__global__ void bias_relu_ln(const float* __restrict__ in, const float* __restrict__ bias,
                             const float* __restrict__ g, const float* __restrict__ be,
                             const float* __restrict__ res, int resStride,
                             float* __restrict__ out, int F, int pt) {
  __shared__ float red[2][8];
  int n = blockIdx.x, t = threadIdx.x, BT = blockDim.x;
  const float* ip = in + (size_t)n * F;
  float loc[4];
  float s = 0.f, s2 = 0.f;
  for (int i = 0; i < pt; ++i) {
    int f = t + i * BT;
    float v = ip[f] + bias[f];
    v = (v > 0.f) ? v : 0.f;
    loc[i] = v; s += v; s2 += v * v;
  }
  for (int off = 16; off; off >>= 1) {
    s  += __shfl_down(s,  off, 32);
    s2 += __shfl_down(s2, off, 32);
  }
  int wv = t >> 5, lane = t & 31, nw = (BT + 31) >> 5;
  if (lane == 0) { red[0][wv] = s; red[1][wv] = s2; }
  __syncthreads();
  if (t == 0) {
    float a = 0.f, b = 0.f;
    for (int i = 0; i < nw; ++i) { a += red[0][i]; b += red[1][i]; }
    red[0][0] = a; red[1][0] = b;
  }
  __syncthreads();
  float mu   = red[0][0] / F;
  float var  = red[1][0] / F - mu * mu;
  float rstd = rsqrtf(var + 1e-5f);
  for (int i = 0; i < pt; ++i) {
    int f = t + i * BT;
    float v = (loc[i] - mu) * rstd * g[f] + be[f];
    if (res) v += res[(size_t)n * resStride + f];
    out[(size_t)n * F + f] = v;
  }
}

// ---------------------------------------------------------------------------
// final: out[n] = sigmoid(dot(s[n,0:32], w) + b)
// ---------------------------------------------------------------------------
__global__ void scorer_final(const float* __restrict__ s, const float* __restrict__ w,
                             const float* __restrict__ b, float* __restrict__ out, int N) {
  int n = blockIdx.x * blockDim.x + threadIdx.x;
  if (n >= N) return;
  float acc = b[0];
  #pragma unroll
  for (int k = 0; k < 32; ++k) acc += s[(size_t)n * 32 + k] * w[k];
  out[n] = 1.f / (1.f + __expf(-acc));
}

// ---------------------------------------------------------------------------
// Host-side helpers
// ---------------------------------------------------------------------------
static inline void zero_buf(void* p, size_t nwords, hipStream_t s) {
  unsigned blocks = (unsigned)((nwords + 255) / 256);
  kfill0<<<blocks, 256, 0, s>>>((unsigned*)p, (long)nwords);
}

static void run_gemm(const float* A, const float* B, float* C,
                     int M, int K, int Nc, hipStream_t s) {
  dim3 grid(M / TM, (Nc + TNB - 1) / TNB);
  gemm_wmma_f16<<<grid, 256, 0, s>>>(A, B, C, M, K, Nc);
}

static int ilog2i(int v) { int r = 0; while ((1 << r) < v) ++r; return r; }

static void gat_layer(const float* hin, int Fin, const float* W,
                      const float* asrc, const float* adst,
                      const float* bias, const float* g, const float* be,
                      const float* res, int resStride,
                      int H, int C, float* pre, float* agg, float* hout,
                      float* es, float* ed, unsigned* mk, float* zb, float* ebuf,
                      const int* ei, hipStream_t s) {
  const int F = H * C;
  run_gemm(hin, W, pre, NN, Fin, F, s);
  attn_scores<<<NN, H * 32, 0, s>>>(pre, asrc, adst, es, ed, H, C);
  zero_buf(mk, (size_t)NN * H, s);
  zero_buf(zb, (size_t)NN * H, s);
  const int Etot = EE + NN;
  const unsigned ebks = (Etot + 255) / 256;
  edge_pass1<<<ebks, 256, 0, s>>>(ei, EE, NN, es, ed, ebuf, mk, H);
  edge_pass2<<<ebks, 256, 0, s>>>(ei, EE, NN, ebuf, mk, zb, H);
  edge_alpha<<<ebks, 256, 0, s>>>(ei, EE, NN, zb, ebuf, H);
  zero_buf(agg, (size_t)NN * F, s);
  size_t total = (size_t)Etot * F;
  unsigned abks = (unsigned)((total + 255) / 256);
  edge_agg<<<abks, 256, 0, s>>>(ei, EE, NN, ebuf, pre, agg, H, ilog2i(C), ilog2i(F));
  int BT = (F >= 256) ? 256 : F;
  bias_relu_ln<<<NN, BT, 0, s>>>(agg, bias, g, be, res, resStride, hout, F, F / BT);
}

extern "C" void kernel_launch(void* const* d_in, const int* in_sizes, int n_in,
                              void* d_out, int out_size, void* d_ws, size_t ws_size,
                              hipStream_t stream) {
  const float* x   = (const float*)d_in[0];
  const int*   ei  = (const int*)d_in[1];
  const float* w1  = (const float*)d_in[2];
  const float* as1 = (const float*)d_in[3];
  const float* ad1 = (const float*)d_in[4];
  const float* b1  = (const float*)d_in[5];
  const float* g1  = (const float*)d_in[6];
  const float* be1 = (const float*)d_in[7];
  const float* w2  = (const float*)d_in[8];
  const float* as2 = (const float*)d_in[9];
  const float* ad2 = (const float*)d_in[10];
  const float* b2  = (const float*)d_in[11];
  const float* g2  = (const float*)d_in[12];
  const float* be2 = (const float*)d_in[13];
  const float* w3  = (const float*)d_in[14];
  const float* as3 = (const float*)d_in[15];
  const float* ad3 = (const float*)d_in[16];
  const float* b3  = (const float*)d_in[17];
  const float* g3  = (const float*)d_in[18];
  const float* be3 = (const float*)d_in[19];
  const float* sw1 = (const float*)d_in[20];
  const float* sb1 = (const float*)d_in[21];
  const float* sg1 = (const float*)d_in[22];
  const float* sbe1= (const float*)d_in[23];
  const float* sw2 = (const float*)d_in[24];
  const float* sb2 = (const float*)d_in[25];
  const float* sg2 = (const float*)d_in[26];
  const float* sbe2= (const float*)d_in[27];
  const float* sw3 = (const float*)d_in[28];
  const float* sb3 = (const float*)d_in[29];
  float* out = (float*)d_out;

  // ---- workspace carve ----
  size_t off = 0;
  char* base = (char*)d_ws;
  auto carve = [&](size_t bytes) -> void* {
    void* p = base + off;
    off += (bytes + 255) & ~(size_t)255;
    return p;
  };
  float*    pre  = (float*)   carve((size_t)NN * 1024 * 4);
  float*    agg  = (float*)   carve((size_t)NN * 1024 * 4);
  float*    h1   = (float*)   carve((size_t)NN * 1024 * 4);
  float*    h2   = (float*)   carve((size_t)NN * 256 * 4);
  float*    h3   = (float*)   carve((size_t)NN * 64 * 4);
  float*    es   = (float*)   carve((size_t)NN * 8 * 4);
  float*    ed   = (float*)   carve((size_t)NN * 8 * 4);
  unsigned* mk   = (unsigned*)carve((size_t)NN * 8 * 4);
  float*    zb   = (float*)   carve((size_t)NN * 8 * 4);
  float*    ebuf = (float*)   carve((size_t)(EE + NN) * 8 * 4);
  (void)ws_size; (void)n_in; (void)in_sizes; (void)out_size;

  // ---- GAT layer 1: 512 -> 8 x 128 ----
  gat_layer(x, DD, w1, as1, ad1, b1, g1, be1, nullptr, 0,
            8, 128, pre, agg, h1, es, ed, mk, zb, ebuf, ei, stream);
  // ---- GAT layer 2: 1024 -> 4 x 64, residual h1[:, :256] ----
  gat_layer(h1, 1024, w2, as2, ad2, b2, g2, be2, h1, 1024,
            4, 64, pre, agg, h2, es, ed, mk, zb, ebuf, ei, stream);
  // ---- GAT layer 3: 256 -> 2 x 32, residual h2[:, :64] ----
  gat_layer(h2, 256, w3, as3, ad3, b3, g3, be3, h2, 256,
            2, 32, pre, agg, h3, es, ed, mk, zb, ebuf, ei, stream);

  // ---- scorer MLP ----
  // s1 = LN(relu(h3 @ sw1 + sb1))
  run_gemm(h3, sw1, pre, NN, 64, 64, stream);
  bias_relu_ln<<<NN, 64, 0, stream>>>(pre, sb1, sg1, sbe1, nullptr, 0, agg, 64, 1);
  // s2 = LN(relu(s1 @ sw2 + sb2))
  run_gemm(agg, sw2, pre, NN, 64, 32, stream);
  bias_relu_ln<<<NN, 32, 0, stream>>>(pre, sb2, sg2, sbe2, nullptr, 0, agg, 32, 1);
  // out = sigmoid(s2 @ sw3 + sb3)
  scorer_final<<<(NN + 255) / 256, 256, 0, stream>>>(agg, sw3, sb3, out, NN);
}
